// HeadLoss_20658792694162
// MI455X (gfx1250) — matrix operand order repaired and checked by reading, not compiled, not run on hardware
//
#include <hip/hip_runtime.h>

// ---------------------------------------------------------------------------
// MI455X / gfx1250: masked Gram (H^T diag(mask_k) H) for 10 classes via
// v_wmma_f32_16x16x32_f16, then Cholesky logdet on the 10 SPD 64x64 matrices.
//
// Roofline: read h once = 128 MB -> 5.5 us floor @ 23.3 TB/s. Ideal compute
// 4.1 GFLOP; the 10-class A-mask redundancy makes it 41 GFLOP, which f16 WMMA
// (multi-PFLOP dense) absorbs near the memory floor. This revision slashes the
// mask-building VALU overhead (was ~47 VALU per WMMA) by reusing one masked A
// fragment across 4 B tiles and using a one-hot + (t<<16)-t mask expansion.
// ---------------------------------------------------------------------------

typedef __attribute__((ext_vector_type(16))) _Float16 v16h;
typedef __attribute__((ext_vector_type(8)))  float    v8f;
typedef __attribute__((ext_vector_type(8)))  unsigned v8u;

#define NROWS   500000
#define DIM     64
#define NCLS    10
#define CHUNK   32
#define NCHUNKS (NROWS / CHUNK)          // 15625 (exact)
#define GRAM_ELEMS (NCLS * DIM * DIM)    // 40960 floats

// ---------------------------------------------------------------------------
// Zero scratch (graph-capture safe; no hipMemset).
// ---------------------------------------------------------------------------
__global__ void zero_kernel(float* __restrict__ gram, int* __restrict__ counts) {
  int i = blockIdx.x * blockDim.x + threadIdx.x;
  if (i < GRAM_ELEMS) gram[i] = 0.0f;
  if (i < NCLS) counts[i] = 0;
}

// ---------------------------------------------------------------------------
// Gram kernel: 128 threads = 4 waves. Wave w owns output row-block a = w and
// all four column-blocks b = 0..3 of the 4x4 tiling of the 64x64 Gram. Per
// 32-row chunk and per class, ONE masked A fragment feeds FOUR WMMAs.
// ---------------------------------------------------------------------------
__global__ __launch_bounds__(128) void gram_kernel(const float* __restrict__ h,
                                                   const int*   __restrict__ yhat,
                                                   float*       __restrict__ gram,
                                                   int*         __restrict__ counts) {
  __shared__ _Float16       ldsT[DIM * CHUNK];   // [col][row] f16 (col-major in rows)
  __shared__ unsigned short ycls[CHUNK];         // class id per chunk row
  __shared__ int            lcnt[NCLS];

  const int tid     = threadIdx.x;
  const int lane    = tid & 31;
  const int wave    = tid >> 5;            // 0..3 == output row-block a
  const int half_id = (lane >> 4) & 1;     // 0: lanes 0-15, 1: lanes 16-31
  const int ln16    = lane & 15;

  if (tid < NCLS) lcnt[tid] = 0;

  // 4 column-blocks x 10 classes of f32 16x16 accumulators (v8f each)
  v8f acc[4][NCLS];
  const v8f vzero = {};
  #pragma unroll
  for (int b = 0; b < 4; ++b)
    #pragma unroll
    for (int c = 0; c < NCLS; ++c)
      acc[b][c] = vzero;

  __syncthreads();

  for (int chunk = blockIdx.x; chunk < NCHUNKS; chunk += gridDim.x) {
    // Prefetch next grid-stride chunk (global_prefetch_b8 on gfx1250).
    int nxt = chunk + gridDim.x;
    if (nxt < NCHUNKS)
      __builtin_prefetch(h + (long)nxt * (CHUNK * DIM) + tid * 16, 0, 1);

    // ---- load 32 rows x 64 cols f32, convert to f16, store transposed ----
    const float4* hv = (const float4*)(h + (long)chunk * (CHUNK * DIM));
    #pragma unroll
    for (int it = 0; it < 4; ++it) {
      int vidx = tid * 4 + it;             // 0..511 float4s
      float4 f = hv[vidx];
      int e   = vidx * 4;
      int row = e >> 6;                    // 0..31
      int col = e & 63;                    // 0..63 (multiple of 4)
      ldsT[(col + 0) * CHUNK + row] = (_Float16)f.x;
      ldsT[(col + 1) * CHUNK + row] = (_Float16)f.y;
      ldsT[(col + 2) * CHUNK + row] = (_Float16)f.z;
      ldsT[(col + 3) * CHUNK + row] = (_Float16)f.w;
    }
    if (tid < CHUNK) {
      int yi = yhat[chunk * CHUNK + tid];
      ycls[tid] = (unsigned short)yi;
      atomicAdd(&lcnt[yi], 1);             // LDS integer atomic (deterministic)
    }
    __syncthreads();

    const unsigned* l32 = (const unsigned*)ldsT;   // packed f16 pairs

    // ---- A fragment (row-block a = wave) + one-hot class words, once/chunk
    // 16-bit A 16x32 layout: lane = M; VGPR v holds K pair:
    //   ka = (v<4 ? 0 : 16) + 8*half + 2*(v&3)
    v8u au, oh;
    #pragma unroll
    for (int v = 0; v < 8; ++v) {
      int ka   = ((v < 4) ? 0 : 16) + half_id * 8 + (v & 3) * 2;
      int colA = wave * 16 + ln16;
      au[v] = l32[colA * (CHUNK / 2) + (ka >> 1)];
      oh[v] = (1u << ycls[ka]) | ((unsigned)(1u << ycls[ka + 1]) << 16);
    }

    // ---- B fragments for all four column-blocks
    // 16-bit B 32x16 layout (same K order as A): kb = (v<4 ? 0 : 16) + 8*half + 2*(v&3)
    v8u bu[4];
    #pragma unroll
    for (int b = 0; b < 4; ++b) {
      #pragma unroll
      for (int v = 0; v < 8; ++v) {
        int kb   = ((v < 4) ? 0 : 16) + half_id * 8 + (v & 3) * 2;
        int colB = b * 16 + ln16;
        bu[b][v] = l32[colB * (CHUNK / 2) + (kb >> 1)];
      }
    }

    // ---- per class: one masked A fragment -> 4 WMMAs
    #pragma unroll
    for (int c = 0; c < NCLS; ++c) {
      v8u am;
      #pragma unroll
      for (int v = 0; v < 8; ++v) {
        unsigned t = (oh[v] >> c) & 0x00010001u;   // per-halfword "cls==c" bit
        am[v] = au[v] & ((t << 16) - t);           // expand bit -> 0xFFFF mask
      }
      const v16h av = __builtin_bit_cast(v16h, am);
      #pragma unroll
      for (int b = 0; b < 4; ++b) {
        acc[b][c] = __builtin_amdgcn_wmma_f32_16x16x32_f16(
            /*neg_a=*/false, av,
            /*neg_b=*/false, __builtin_bit_cast(v16h, bu[b]),
            /*c_mod=*/(short)0, acc[b][c],
            /*reuse_a=*/false, /*reuse_b=*/false);
      }
    }
    __syncthreads();   // protect LDS before next chunk overwrites it
  }

  // ---- reduce partial Grams into global scratch ----
  // C/D f32 16x16 layout: VGPR r -> M = r + 8*half, N = lane%16.
  #pragma unroll
  for (int b = 0; b < 4; ++b) {
    #pragma unroll
    for (int c = 0; c < NCLS; ++c) {
      #pragma unroll
      for (int r = 0; r < 8; ++r) {
        int m = wave * 16 + r + 8 * half_id;
        int n = b * 16 + ln16;
        atomicAdd(&gram[c * (DIM * DIM) + m * DIM + n], acc[b][c][r]);
      }
    }
  }
  if (tid < NCLS) atomicAdd(&counts[tid], lcnt[tid]);
}

// ---------------------------------------------------------------------------
// logdet via in-LDS Cholesky of M = G*0.5/cnt + I (SPD). One block, 64 threads
// (thread i owns row i). 0.5*logdet(M) = sum_i log L_ii. Output: f32 scalar.
// ---------------------------------------------------------------------------
__global__ __launch_bounds__(64) void logdet_kernel(const float* __restrict__ gram,
                                                    const int*   __restrict__ counts,
                                                    float*       __restrict__ out) {
  __shared__ float A[DIM][DIM + 1];
  const int tid = threadIdx.x;   // 0..63
  float total = 0.0f;            // only thread 0's copy is used

  for (int c = 0; c < NCLS; ++c) {
    float inv = 0.5f / (float)counts[c];
    for (int j = 0; j < DIM; ++j)
      A[tid][j] = gram[c * DIM * DIM + tid * DIM + j] * inv + (tid == j ? 1.0f : 0.0f);
    __syncthreads();

    for (int j = 0; j < DIM; ++j) {
      if (tid == j) {
        float s = A[j][j];
        for (int k = 0; k < j; ++k) s -= A[j][k] * A[j][k];
        A[j][j] = sqrtf(s);
      }
      __syncthreads();
      if (tid > j) {
        float s = A[tid][j];
        for (int k = 0; k < j; ++k) s -= A[tid][k] * A[j][k];
        A[tid][j] = s / A[j][j];
      }
      __syncthreads();
    }

    if (tid == 0) {
      float s = 0.0f;
      for (int i = 0; i < DIM; ++i) s += logf(A[i][i]);
      total += s;
    }
    __syncthreads();
  }

  if (tid == 0) out[0] = total;
}

// ---------------------------------------------------------------------------
extern "C" void kernel_launch(void* const* d_in, const int* in_sizes, int n_in,
                              void* d_out, int out_size, void* d_ws, size_t ws_size,
                              hipStream_t stream) {
  (void)in_sizes; (void)n_in; (void)out_size; (void)ws_size;
  const float* h    = (const float*)d_in[0];
  const int*   yhat = (const int*)d_in[1];

  float* gram   = (float*)d_ws;                                     // 10*64*64 f32
  int*   counts = (int*)((char*)d_ws + GRAM_ELEMS * sizeof(float)); // 10 ints

  zero_kernel<<<(GRAM_ELEMS + 255) / 256, 256, 0, stream>>>(gram, counts);
  gram_kernel<<<512, 128, 0, stream>>>(h, yhat, gram, counts);
  logdet_kernel<<<1, 64, 0, stream>>>(gram, counts, (float*)d_out);
}
